// DBNN_59545426591795
// MI455X (gfx1250) — matrix operand order repaired and checked by reading, not compiled, not run on hardware
//
#include <hip/hip_runtime.h>
#include <hip/hip_bf16.h>

// MI455X / gfx1250, wave32. Two WMMA GEMM stages:
//  1) causal depthwise conv as Toeplitz(k_n)[T,T] x X[T,B] per dimension n
//     (A-operand served by aligned b128 loads from 8 shift-copies of the
//      element-reversed kernel)
//  2) bilinear form via Z = Y * Wm^T per (batch, t-tile) + butterfly reduction
typedef __attribute__((ext_vector_type(16))) __bf16 v16bf;
typedef __attribute__((ext_vector_type(8)))  float  v8f;

#define T_DIM 1024
#define N_DIM 256
#define B_DIM 32
#define KSR_LEN 1056   // covers qa+23 <= 1054, multiple of 8

static __device__ __forceinline__ unsigned short f2bf(float f) {
  union { float f; unsigned u; } c; c.f = f;
  unsigned u = c.u;
  u += 0x7FFFu + ((u >> 16) & 1u);       // round-to-nearest-even
  return (unsigned short)(u >> 16);
}
static __device__ __forceinline__ float bf2f(unsigned short s) {
  union { unsigned u; float f; } c; c.u = ((unsigned)s) << 16;
  return c.f;
}

union ABFrag { v16bf v; unsigned short s[16]; uint4 q[2]; };

// ---------------------------------------------------------------------------
// Prep: Wm = W with diagonal zeroed, converted to bf16 (B-operand of stage 2)
// ---------------------------------------------------------------------------
__global__ void prep_w_kernel(const float* __restrict__ W,
                              unsigned short* __restrict__ Wbf) {
  const int n = blockIdx.x;
  const int m = threadIdx.x;
  const float w = W[n * N_DIM + m];
  Wbf[n * N_DIM + m] = (n == m) ? (unsigned short)0 : f2bf(w);
}

// ---------------------------------------------------------------------------
// Stage 1: per-dimension causal conv as Toeplitz GEMM.
// Block = one n (256 blocks, 8 waves). LDS: bf16 x for 16 batches (32KB),
// kernel (2KB), 8 shifted reversed-kernel copies (17KB).
// Output written as yt[b, t, n] in bf16.
// ---------------------------------------------------------------------------
__global__ void conv_wmma_kernel(const float* __restrict__ x,
                                 const float* __restrict__ tau_rise,
                                 const float* __restrict__ tau_decay,
                                 const float* __restrict__ omega,
                                 unsigned short* __restrict__ ytbf) {
  __shared__ __align__(16) unsigned short xs[16 * T_DIM];  // x[b,n,:] bf16
  __shared__ unsigned short ks[T_DIM];                     // k[n,d] bf16
  __shared__ __align__(16) unsigned short ksrs[8 * KSR_LEN];

  const int n    = blockIdx.x;
  const int tid  = threadIdx.x;
  const int lane = tid & 31;
  const int wave = tid >> 5;
  const int hi   = (lane >> 4) & 1;
  const int row  = lane & 15;

  // k[n,d] = om*(1-exp(-d/tr))*exp(-d/td)
  const float tr = tau_rise[n], td = tau_decay[n], om = omega[n];
  for (int d = tid; d < T_DIM; d += 256) {
    const float t = (float)d;
    ks[d] = f2bf(om * (1.0f - __expf(-t / tr)) * __expf(-t / td));
  }
  __syncthreads();

  // 8 shifted copies of the element-reversed padded kernel:
  // ksrs[p][i] = ksr[i+p], ksr[j] = (j<=1023) ? k[1023-j] : 0
  for (int i = tid; i < 8 * KSR_LEN; i += 256) {
    const int p = i / KSR_LEN;
    const int j = (i - p * KSR_LEN) + p;
    ksrs[i] = (j <= T_DIM - 1) ? ks[T_DIM - 1 - j] : (unsigned short)0;
  }

  const int p = (T_DIM - 1 - row) & 7;               // per-lane fixed shift
  const unsigned short* ksl = ksrs + p * KSR_LEN;

  for (int bh = 0; bh < 2; ++bh) {            // batch halves: b in [16*bh, +16)
    __syncthreads();
    for (int i = tid; i < 16 * T_DIM; i += 256) {
      const int b = (i >> 10), s = i & (T_DIM - 1);
      xs[i] = f2bf(x[((bh * 16 + b) * N_DIM + n) * T_DIM + s]);
    }
    __syncthreads();

    for (int tt = wave; tt < T_DIM / 16; tt += 8) {
      const int t0 = tt * 16;
      // aligned base so that ksl[q0 + s0 + e] == ksr[q + e], q = 1023-t+s0+8hi
      const int q0 = (T_DIM - 1) - t0 - row + 8 * hi - p;
      v8f c = {};
      for (int s0 = 0; s0 < t0 + 16; s0 += 32) {   // triangular K loop
        // A: 16x32 Toeplitz tile, two aligned b128 runs from reversed kernel
        ABFrag a;
        const uint4* pa = (const uint4*)(ksl + q0 + s0);
        a.q[0] = pa[0];          // elements 0..7  -> ksr[q .. q+7]
        a.q[1] = pa[2];          // elements 8..15 -> ksr[q+16 .. q+23]
        // B: 32x16 x tile, col = batch (lane&15), K runs s0+8hi.. and s0+16+8hi..
        ABFrag bfr;
        const uint4* pb = (const uint4*)(xs + row * T_DIM + s0 + 8 * hi);
        bfr.q[0] = pb[0];
        bfr.q[1] = pb[2];
        c = __builtin_amdgcn_wmma_f32_16x16x32_bf16(false, a.v, false, bfr.v,
                                                    (short)0, c, false, false);
      }
      // D layout: t = t0 + r + 8*hi, batch col = row. Store transposed yt[b,t,n].
#pragma unroll
      for (int r = 0; r < 8; ++r) {
        const int tw = t0 + r + 8 * hi;
        const int b  = bh * 16 + row;
        ytbf[(b * T_DIM + tw) * N_DIM + n] = f2bf(c[r]);
      }
    }
  }
}

// ---------------------------------------------------------------------------
// Stage 2: out[b,t] = y^T Wm y + sum(y) - 70, via Z = Y x Wm^T (WMMA) then
// a shuffle-butterfly dot. Block = (t-tile, batch), 8 waves x 2 n-tiles each.
// ---------------------------------------------------------------------------
__global__ void bilinear_wmma_kernel(const unsigned short* __restrict__ ytbf,
                                     const unsigned short* __restrict__ Wbf,
                                     float* __restrict__ out) {
  __shared__ float sAcc[16];
  const int tt   = blockIdx.x;                 // 0..63
  const int b    = blockIdx.y;                 // 0..31
  const int t0   = tt * 16;
  const int tid  = threadIdx.x;
  const int lane = tid & 31;
  const int wave = tid >> 5;
  const int hi   = (lane >> 4) & 1;
  const int row  = lane & 15;

  if (tid < 16) sAcc[tid] = 0.0f;
  __syncthreads();

  const unsigned short* ybase = ytbf + (size_t)b * T_DIM * N_DIM;
  const int n0a = wave * 16;
  const int n0b = (wave + 8) * 16;
  const int t   = t0 + row;

  v8f acc0 = {}; v8f acc1 = {};
  float lsum = 0.0f;                           // linear term Sum_m y[t,m] (wave 0)

  for (int m0 = 0; m0 < N_DIM; m0 += 32) {
    // A: Y tile 16x32, two contiguous 8-element runs per lane
    ABFrag a;
    const unsigned short* ar = ybase + t * N_DIM + m0 + 16 * hi;
    a.q[0] = *(const uint4*)(ar);
    a.q[1] = *(const uint4*)(ar + 8);
    if (wave == 0) {                           // each (t,m) appears once per wave
#pragma unroll
      for (int e = 0; e < 16; ++e) lsum += bf2f(a.s[e]);
    }
    // B: Wm^T tiles, B[kk, col] = Wm[n0+col, m0+kk], contiguous 16 shorts
    ABFrag w0, w1;
    const int mo = m0 + 16 * hi;
    const uint4* pw0 = (const uint4*)(Wbf + (n0a + row) * N_DIM + mo);
    const uint4* pw1 = (const uint4*)(Wbf + (n0b + row) * N_DIM + mo);
    w0.q[0] = pw0[0]; w0.q[1] = pw0[1];
    w1.q[0] = pw1[0]; w1.q[1] = pw1[1];
    acc0 = __builtin_amdgcn_wmma_f32_16x16x32_bf16(false, a.v, false, w0.v,
                                                   (short)0, acc0, false, false);
    acc1 = __builtin_amdgcn_wmma_f32_16x16x32_bf16(false, a.v, false, w1.v,
                                                   (short)0, acc1, false, false);
  }

  if (wave == 0) {                             // fold lanes L and L+16 (same t)
    lsum += __shfl_xor(lsum, 16, 32);
    if (lane < 16) atomicAdd(&sAcc[lane], lsum);
  }

  // bilinear dot: Sum_n Z[t,n] * y[t,n], Z at (t = t0+r+8*hi, n = n0+row)
#pragma unroll
  for (int r = 0; r < 8; ++r) {
    const int tz = t0 + r + 8 * hi;
    float p = acc0[r] * bf2f(ybase[tz * N_DIM + n0a + row])
            + acc1[r] * bf2f(ybase[tz * N_DIM + n0b + row]);
    p += __shfl_xor(p, 1, 32);
    p += __shfl_xor(p, 2, 32);
    p += __shfl_xor(p, 4, 32);
    p += __shfl_xor(p, 8, 32);
    if (row == 0) atomicAdd(&sAcc[r + 8 * hi], p);
  }

  __syncthreads();
  if (tid < 16) out[b * T_DIM + t0 + tid] = sAcc[tid] - 70.0f;
}

// ---------------------------------------------------------------------------
extern "C" void kernel_launch(void* const* d_in, const int* in_sizes, int n_in,
                              void* d_out, int out_size, void* d_ws, size_t ws_size,
                              hipStream_t stream) {
  const float* x  = (const float*)d_in[0];
  const float* tr = (const float*)d_in[1];
  const float* td = (const float*)d_in[2];
  const float* om = (const float*)d_in[3];
  const float* W  = (const float*)d_in[4];
  float* out = (float*)d_out;

  // workspace layout: yt bf16 [B,T,N] (16 MiB) then Wm bf16 [N,N] (128 KiB)
  unsigned short* ytbf = (unsigned short*)d_ws;
  unsigned short* Wbf  =
      (unsigned short*)((char*)d_ws + (size_t)B_DIM * T_DIM * N_DIM * sizeof(unsigned short));

  prep_w_kernel<<<dim3(N_DIM), dim3(N_DIM), 0, stream>>>(W, Wbf);
  conv_wmma_kernel<<<dim3(N_DIM), dim3(256), 0, stream>>>(x, tr, td, om, ytbf);
  bilinear_wmma_kernel<<<dim3(T_DIM / 16, B_DIM), dim3(256), 0, stream>>>(ytbf, Wbf, out);
}
